// DWTFeatureModel_50783693308122
// MI455X (gfx1250) — matrix operand order repaired and checked by reading, not compiled, not run on hardware
//
#include <hip/hip_runtime.h>
#include <hip/hip_bf16.h>

// ---------------- problem constants (from reference) ----------------
#define SW      512
#define SUBW    128
#define NBINS   4          // SW / SUBW
#define NF      48
#define TLEN    154        // DWT_OUT_LEN
#define KDIM    (16 * TLEN)   // 2464 = (h*w=16) * 154, divisible by 32 (77 chunks)
#define NEGSL   0.02f

// db4 analysis filters (DEC_LO / DEC_HI); reference uses reversed filters in
// correlation, i.e. y[k] = sum_j x[2k-6+j] * DEC[7-j]
__constant__ float C_LO[8] = {
    -0.010597401784997278f,  0.032883011666982945f,  0.030841381835986965f,
    -0.18703481171888114f,  -0.02798376941698385f,   0.6308807679295904f,
     0.7148465705525415f,    0.23037781330885523f };
__constant__ float C_HI[8] = {
    -0.23037781330885523f,   0.7148465705525415f,   -0.6308807679295904f,
    -0.02798376941698385f,   0.18703481171888114f,   0.030841381835986965f,
    -0.032883011666982945f, -0.010597401784997278f };

// concat offsets: [lo4(14), hi1(67), hi2(37), hi3(22), hi4(14)]
#define OFF_LO4  0
#define OFF_HI1  14
#define OFF_HI2  81
#define OFF_HI3  118
#define OFF_HI4  140

// ---------------- kernel 1: maxpool + 4-level DWT ----------------
// one block per sample; 64 sequences of length 128 live in LDS.
#define PA 129   // odd pitch -> LDS bank-conflict free across sequences
#define PB 69

template<int NN, int NOUT, int PIN, int POUT>
__device__ inline void dwt_level(const float* __restrict__ in,
                                 float* __restrict__ loLds,
                                 _Float16* __restrict__ cb,   // coeff + n*64*TLEN
                                 int hiOff, int loOff,        // loOff >= 0 => lo also to global
                                 int tid)
{
    for (int i = tid; i < 64 * NOUT; i += 256) {
        int s = i / NOUT;
        int k = i - s * NOUT;
        const float* row = in + s * PIN;
        float lo = 0.f, hi = 0.f;
#pragma unroll
        for (int j = 0; j < 8; ++j) {
            int q = 2 * k - 6 + j;            // left reflect pad = 6 at every level
            q = (q < 0) ? -q : q;             // reflect (no edge repeat)
            q = (q >= NN) ? (2 * (NN - 1) - q) : q;
            float v = row[q];
            lo = fmaf(v, C_LO[7 - j], lo);
            hi = fmaf(v, C_HI[7 - j], hi);
        }
        cb[s * TLEN + hiOff + k] = (_Float16)hi;
        if (loOff >= 0) cb[s * TLEN + loOff + k] = (_Float16)lo;
        else            loLds[s * POUT + k] = lo;
    }
}

__global__ __launch_bounds__(256)
void k_pool_dwt(const float* __restrict__ x, _Float16* __restrict__ coeff)
{
    __shared__ float bufA[64 * PA];
    __shared__ float bufB[64 * PB];

    const int n   = blockIdx.x;
    const int tid = threadIdx.x;
    const float* xb = x + (size_t)n * (SW * 64);
    _Float16* cb = coeff + (size_t)n * 64 * TLEN;

    // 2x2 max-pool of the 8x8 tail; bufA[s][t], s = bin*16 + h*4 + w
    for (int p = tid; p < SW * 16; p += 256) {
        int tg   = p >> 4;
        int sidx = p & 15;
        int h = sidx >> 2, w = sidx & 3;
        const float* tp = xb + (size_t)tg * 64 + (h * 16 + w * 2);
        float2 r0 = *(const float2*)(tp);
        float2 r1 = *(const float2*)(tp + 8);
        float m = fmaxf(fmaxf(r0.x, r0.y), fmaxf(r1.x, r1.y));
        int bin = tg >> 7;
        int t   = tg & 127;
        bufA[(bin * 16 + sidx) * PA + t] = m;
    }
    __syncthreads();

    // level 1: 128 -> 67    (lo -> bufB, hi -> global @14)
    dwt_level<128, 67, PA, PB>(bufA, bufB, cb, OFF_HI1, -1, tid);
    __syncthreads();
    // level 2: 67 -> 37     (lo -> bufA, hi -> global @81)
    dwt_level<67, 37, PB, PA>(bufB, bufA, cb, OFF_HI2, -1, tid);
    __syncthreads();
    // level 3: 37 -> 22     (lo -> bufB, hi -> global @118)
    dwt_level<37, 22, PA, PB>(bufA, bufB, cb, OFF_HI3, -1, tid);
    __syncthreads();
    // level 4: 22 -> 14     (lo -> global @0, hi -> global @140)
    dwt_level<22, 14, PB, PA>(bufB, bufA, cb, OFF_HI4, OFF_LO4, tid);
}

// ---------------- kernel 2: conv_w (b,f,t,h,w) -> f16 (b,f, hw*154+t) ----------------
__global__ __launch_bounds__(256)
void k_wt(const float* __restrict__ w, _Float16* __restrict__ wt, int total)
{
    int i = blockIdx.x * blockDim.x + threadIdx.x;
    if (i >= total) return;
    int k  = i % KDIM;
    int bf = i / KDIM;               // b*48 + f
    int hw = k / TLEN;
    int t  = k - hw * TLEN;
    wt[(size_t)bf * KDIM + k] = (_Float16)w[((size_t)bf * TLEN + t) * 16 + hw];
}

// ---------------- kernel 3: WMMA GEMM per bin ----------------
typedef _Float16 v16h __attribute__((ext_vector_type(16)));
typedef float    v8f  __attribute__((ext_vector_type(8)));

__global__ __launch_bounds__(128)
void k_gemm(const _Float16* __restrict__ coeff,
            const _Float16* __restrict__ wt,
            const float*    __restrict__ bias,
            float*          __restrict__ out,
            int n)
{
    const int lane  = threadIdx.x & 31;
    const int wave  = threadIdx.x >> 5;
    const int mTile = blockIdx.x * 4 + wave;   // 16 output rows per wave
    const int b     = blockIdx.y;              // bin

    const int l16    = lane & 15;
    const int hiHalf = (lane >> 4) & 1;
    const int kbA    = hiHalf * 16;            // A: K group  [kb..kb+15]
    const int kbB    = hiHalf * 16;            // B: K group  [kb2..kb2+15]

    const int rowA = mTile * 16 + l16;         // A-fragment row (M = lane&15, both halves)
    const _Float16* aBase = coeff + ((size_t)rowA * NBINS + b) * KDIM;
    const _Float16* b0 = wt + ((size_t)(b * NF) + 0  + l16) * KDIM;
    const _Float16* b1 = wt + ((size_t)(b * NF) + 16 + l16) * KDIM;
    const _Float16* b2 = wt + ((size_t)(b * NF) + 32 + l16) * KDIM;

    v8f acc[3];
    acc[0] = (v8f){}; acc[1] = (v8f){}; acc[2] = (v8f){};

    for (int kc = 0; kc < KDIM; kc += 32) {
        union { v16h v; uint4 u[2]; } a, f0, f1, f2;
        a.u[0]  = *(const uint4*)(aBase + kc + kbA);
        a.u[1]  = *(const uint4*)(aBase + kc + kbA + 8);
        f0.u[0] = *(const uint4*)(b0 + kc + kbB);
        f0.u[1] = *(const uint4*)(b0 + kc + kbB + 8);
        f1.u[0] = *(const uint4*)(b1 + kc + kbB);
        f1.u[1] = *(const uint4*)(b1 + kc + kbB + 8);
        f2.u[0] = *(const uint4*)(b2 + kc + kbB);
        f2.u[1] = *(const uint4*)(b2 + kc + kbB + 8);
        acc[0] = __builtin_amdgcn_wmma_f32_16x16x32_f16(false, a.v, false, f0.v,
                                                        (short)0, acc[0], false, false);
        acc[1] = __builtin_amdgcn_wmma_f32_16x16x32_f16(false, a.v, false, f1.v,
                                                        (short)0, acc[1], false, false);
        acc[2] = __builtin_amdgcn_wmma_f32_16x16x32_f16(false, a.v, false, f2.v,
                                                        (short)0, acc[2], false, false);
    }

    // C/D layout: VGPR r -> M = r + 8*hiHalf, N = lane&15
    const int ncols = NBINS * NF;  // 192
#pragma unroll
    for (int nt = 0; nt < 3; ++nt) {
        int f = nt * 16 + l16;
        float bv = bias[b * NF + f];
#pragma unroll
        for (int r = 0; r < 8; ++r) {
            int m = r + hiHalf * 8;
            float v = acc[nt][r] + bv;
            v = (v >= 0.f) ? v : v * NEGSL;
            out[(size_t)(mTile * 16 + m) * ncols + b * NF + f] = v;
        }
    }
}

// ---------------- launch ----------------
extern "C" void kernel_launch(void* const* d_in, const int* in_sizes, int n_in,
                              void* d_out, int out_size, void* d_ws, size_t ws_size,
                              hipStream_t stream)
{
    const float* x      = (const float*)d_in[0];   // (n,1,512,8,8)
    const float* conv_w = (const float*)d_in[1];   // (4,48,154,4,4)
    const float* conv_b = (const float*)d_in[2];   // (4,48)
    float* out = (float*)d_out;

    const int n = in_sizes[0] / (SW * 64);         // 2048

    _Float16* coeff = (_Float16*)d_ws;
    size_t coeffHalfs = (size_t)n * 64 * TLEN;
    coeffHalfs = (coeffHalfs + 127) & ~(size_t)127;
    _Float16* wt = coeff + coeffHalfs;

    // 1) fused maxpool + 4-level DWT, f16 coeffs to scratch
    k_pool_dwt<<<n, 256, 0, stream>>>(x, coeff);

    // 2) weight repack to f16 with K = hw*154 + t
    const int totalW = NBINS * NF * KDIM;
    k_wt<<<(totalW + 255) / 256, 256, 0, stream>>>(conv_w, wt, totalW);

    // 3) WMMA GEMM: grid (n/64 blocks of 4 waves, 4 bins)
    dim3 g(n / 64, NBINS);
    k_gemm<<<g, 128, 0, stream>>>(coeff, wt, conv_b, out, n);
}
